// MultiHeadAttention_49495203119578
// MI455X (gfx1250) — compile-verified, matrix operand order checked by reading, not run on hardware
//
#include <hip/hip_runtime.h>
#include <hip/hip_bf16.h>

// ---------------------------------------------------------------------------
// Types for gfx1250 WMMA (wave32): V_WMMA_F32_16X16X32_BF16
// ---------------------------------------------------------------------------
typedef __attribute__((ext_vector_type(16))) __bf16        v16bf;
typedef __attribute__((ext_vector_type(8)))  float         v8f;
typedef __attribute__((ext_vector_type(4)))  unsigned int  u32x4;
typedef __attribute__((ext_vector_type(8)))  unsigned int  u32x8;

__device__ __forceinline__ v8f wmma_bf16(v16bf a, v16bf b, v8f c) {
  return __builtin_amdgcn_wmma_f32_16x16x32_bf16(
      false, a, false, b, (short)0, c, false, false);
}

// Fragment load: lane L holds row (L&15), K-chunks (L>>4)*8..+7 and +16..+23.
__device__ __forceinline__ v16bf frag_ld(const __bf16* p) {
  union { u32x4 u[2]; v16bf v; } r;
  r.u[0] = *(const u32x4*)(p);
  r.u[1] = *(const u32x4*)(p + 16);
  return r.v;
}

// ---------------------------------------------------------------------------
// Problem constants
// ---------------------------------------------------------------------------
#define BATCH   4
#define SEQ     2048
#define DMODEL  1024
#define NHEADS  16
#define HDIM    64
#define MTOK    (BATCH * SEQ)          // 8192 tokens
#define D3      (3 * DMODEL)           // 3072

// ---------------------------------------------------------------------------
// TDM 2D tile load: rows x 32 bf16 elements, row stride K=1024 elems in
// memory; LDS receives rows padded 64B+16B -> stride 40 elems (bank-clean).
// D# per CDNA5 ISA ch.8: group0 {count=1, lds_addr, global_addr, type=2},
// group1 {data_size=2B, pad 4dw/16dw, dims/strides, tile dims}.
// 2-operand form: VADDR2/VADDR3 = NULL (<=2D tensor).
// ---------------------------------------------------------------------------
#define GLD 40   // LDS row stride in bf16 elems after TDM padding (80B)

__device__ __forceinline__ void tdm_load_tile(const __bf16* gsrc,
                                              const __bf16* ldst,
                                              unsigned tile_rows) {
  unsigned long long ga = (unsigned long long)gsrc;       // byte address
  unsigned lo = (unsigned)(size_t)ldst;                   // LDS byte offset
  u32x4 g0;
  g0[0] = 1u;                                             // count=1 (valid D#)
  g0[1] = lo;                                             // lds_addr
  g0[2] = (unsigned)ga;                                   // global_addr[31:0]
  g0[3] = ((unsigned)(ga >> 32) & 0x01FFFFFFu) | 0x80000000u; // [56:32]|type=2
  u32x8 g1;
  g1[0] = (1u << 16) | (1u << 20) | (3u << 22) | (3u << 25);
          // data_size=2B | pad_enable | pad_interval=16dw | pad_amount=4dw
  g1[1] = (1024u & 0xFFFFu) << 16;   // tensor_dim0 = 1024 (low16 @ bits 63:48)
  g1[2] = 0xFFFF0000u;               // dim0 hi16=0 | tensor_dim1 low16=65535
  g1[3] = (32u << 16);               // dim1 hi16=0 | tile_dim0 = 32
  g1[4] = tile_rows;                 // tile_dim1 | tile_dim2=0
  g1[5] = 1024u;                     // tensor_dim0_stride low32 (elems)
  g1[6] = 0u;
  g1[7] = 0u;
  asm volatile("tensor_load_to_lds %0, %1" :: "s"(g0), "s"(g1) : "memory");
}

// ---------------------------------------------------------------------------
// Kernel 1a: fp32 -> bf16 elementwise
// ---------------------------------------------------------------------------
__global__ void k_convert_bf16(const float* __restrict__ src,
                               __bf16* __restrict__ dst, int n) {
  int i = blockIdx.x * blockDim.x + threadIdx.x;
  int stride = gridDim.x * blockDim.x;
  for (; i < n; i += stride) dst[i] = (__bf16)src[i];
}

// ---------------------------------------------------------------------------
// Kernel 1b: fp32 [K,N] -> bf16 [N,K]
// ---------------------------------------------------------------------------
__global__ void k_transpose_bf16(const float* __restrict__ src,
                                 __bf16* __restrict__ dst, int K, int N) {
  int i = blockIdx.x * blockDim.x + threadIdx.x;
  int stride = gridDim.x * blockDim.x;
  int total = K * N;
  for (; i < total; i += stride) {
    int k = i / N, n = i - k * N;
    dst[(size_t)n * K + k] = (__bf16)src[i];
  }
}

// ---------------------------------------------------------------------------
// Kernel 2/4: bf16 GEMM, C[M,N] = A[M,K] * Bt[N,K]^T, f32 accumulate.
//  Block tile 128x256, 8 waves (2Mx4N), wave tile 64x64 -> 16 WMMA / K-step.
//  K stepped by 32 with TDM double-buffered LDS staging (DMA overlaps WMMA).
// ---------------------------------------------------------------------------
template <bool F32OUT>
__global__ __launch_bounds__(256)
void k_gemm_bf16(const __bf16* __restrict__ A,
                 const __bf16* __restrict__ Bt,
                 void* __restrict__ Cout,
                 int M, int N, int K) {
  __shared__ __bf16 sA[2][128 * GLD];
  __shared__ __bf16 sB[2][256 * GLD];

  const int tid  = threadIdx.x;
  const int lane = tid & 31;
  const int wave = tid >> 5;
  const int wave_u = __builtin_amdgcn_readfirstlane(tid) >> 5; // SGPR-uniform
  const int wm   = wave >> 2;          // 0..1
  const int wn   = wave & 3;           // 0..3
  const int mBase = blockIdx.x * 128;
  const int nBase = blockIdx.y * 256;

  const int lrow = lane & 15;
  const int lkof = (lane >> 4) << 3;

  v8f acc[4][4];
  {
    v8f z = {};
#pragma unroll
    for (int mi = 0; mi < 4; ++mi)
#pragma unroll
      for (int ni = 0; ni < 4; ++ni) acc[mi][ni] = z;
  }

  const int nk = K / 32;
  int p = 0;

  // Prologue: DMA tile 0 into buffer 0.
  if (wave_u == 0) {
    tdm_load_tile(A  + (size_t)mBase * K, &sA[0][0], 128);
    tdm_load_tile(Bt + (size_t)nBase * K, &sB[0][0], 256);
    __builtin_amdgcn_s_wait_tensorcnt(0);
  }
  __syncthreads();

  for (int kt = 0; kt < nk; ++kt) {
    // Issue DMA for next tile into the other buffer; overlaps compute below.
    if (kt + 1 < nk && wave_u == 0) {
      const int k1 = (kt + 1) * 32;
      tdm_load_tile(A  + (size_t)mBase * K + k1, &sA[p ^ 1][0], 128);
      tdm_load_tile(Bt + (size_t)nBase * K + k1, &sB[p ^ 1][0], 256);
    }

    // Compute on current buffer: 8 fragment loads, 16 WMMAs.
    const __bf16* pA = &sA[p][0];
    const __bf16* pB = &sB[p][0];
    v16bf aF[4], bF[4];
#pragma unroll
    for (int mi = 0; mi < 4; ++mi)
      aF[mi] = frag_ld(&pA[(wm * 64 + mi * 16 + lrow) * GLD + lkof]);
#pragma unroll
    for (int ni = 0; ni < 4; ++ni)
      bF[ni] = frag_ld(&pB[(wn * 64 + ni * 16 + lrow) * GLD + lkof]);
#pragma unroll
    for (int mi = 0; mi < 4; ++mi)
#pragma unroll
      for (int ni = 0; ni < 4; ++ni)
        acc[mi][ni] = wmma_bf16(aF[mi], bF[ni], acc[mi][ni]);

    if (kt + 1 < nk) {
      if (wave_u == 0) __builtin_amdgcn_s_wait_tensorcnt(0);
      __syncthreads();               // publish next buffer / retire reads
      p ^= 1;
    }
  }

  // Straight-line epilogue (templated output type, no per-element branches).
  const int rofs = (lane >> 4) << 3;
#pragma unroll
  for (int mi = 0; mi < 4; ++mi)
#pragma unroll
    for (int ni = 0; ni < 4; ++ni)
#pragma unroll
      for (int g = 0; g < 8; ++g) {
        int row = mBase + wm * 64 + mi * 16 + g + rofs;
        int col = nBase + wn * 64 + ni * 16 + (lane & 15);
        size_t idx = (size_t)row * N + col;
        if constexpr (F32OUT) ((float*)Cout)[idx]  = acc[mi][ni][g];
        else                  ((__bf16*)Cout)[idx] = (__bf16)acc[mi][ni][g];
      }
}

// ---------------------------------------------------------------------------
// Kernel 3: flash attention, causal, bf16 WMMA, f32 online softmax.
// ---------------------------------------------------------------------------
#define LDK 80   // ldsK row stride (32 keys x 64 dims), 160B
#define LDV 40   // ldsVt row stride (64 dims x 32 keys), 80B
#define LDP 48   // per-wave P tile stride, 96B

__global__ __launch_bounds__(256)
void k_flash_attn(const __bf16* __restrict__ qkv,   // [B,S,3072] bf16
                  __bf16* __restrict__ attn) {      // [B,S,1024] bf16
  __shared__ __bf16 ldsK [32 * LDK];
  __shared__ __bf16 ldsVt[64 * LDV];
  __shared__ __bf16 ldsP [8][16 * LDP];

  const int tid  = threadIdx.x;
  const int lane = tid & 31;
  const int wave = tid >> 5;
  const int b  = blockIdx.z;
  const int h  = blockIdx.y;
  const int qb = blockIdx.x;
  const int qt = qb * 128 + wave * 16;

  const int lrow = lane & 15;
  const int lkof = (lane >> 4) << 3;
  const int rofs = (lane >> 4) << 3;

  v16bf qF[2];
#pragma unroll
  for (int c = 0; c < 2; ++c) {
    const __bf16* gq = qkv + ((size_t)b * SEQ + qt + lrow) * D3
                       + h * HDIM + c * 32 + lkof;
    union { u32x4 u[2]; v16bf v; } r;
    r.u[0] = *(const u32x4*)(gq);
    r.u[1] = *(const u32x4*)(gq + 16);
    qF[c] = r.v;
  }

  v8f o[4];
  { v8f z = {}; for (int t = 0; t < 4; ++t) o[t] = z; }
  float m8[8], l8[8];
#pragma unroll
  for (int g = 0; g < 8; ++g) { m8[g] = -3.0e38f; l8[g] = 0.0f; }

  const int nChunks = (qb + 1) * 4;
  const int skey = tid >> 3;
  const int se0  = (tid & 7) * 8;

  for (int kc = 0; kc < nChunks; ++kc) {
    const int kBase = kc * 32;
    const __bf16* gk = qkv + ((size_t)b * SEQ + kBase + skey) * D3
                       + DMODEL + h * HDIM + se0;
    *(u32x4*)&ldsK[skey * LDK + se0] = *(const u32x4*)gk;
    union { u32x4 u; __bf16 e[8]; } vv;
    vv.u = *(const u32x4*)(gk + DMODEL);
#pragma unroll
    for (int j = 0; j < 8; ++j) ldsVt[(se0 + j) * LDV + skey] = vv.e[j];
    __syncthreads();

    v8f s0 = {}, s1 = {};
#pragma unroll
    for (int c = 0; c < 2; ++c) {
      v16bf b0 = frag_ld(&ldsK[(lrow)      * LDK + c * 32 + lkof]);
      v16bf b1 = frag_ld(&ldsK[(16 + lrow) * LDK + c * 32 + lkof]);
      s0 = wmma_bf16(qF[c], b0, s0);
      s1 = wmma_bf16(qF[c], b1, s1);
    }

    const float scale = 0.125f;          // 1/sqrt(64)
    const int colk = lane & 15;
#pragma unroll
    for (int g = 0; g < 8; ++g) {
      const int q = qt + g + rofs;
      float x0 = s0[g] * scale; if (kBase + colk      > q) x0 = -3.0e38f;
      float x1 = s1[g] * scale; if (kBase + 16 + colk > q) x1 = -3.0e38f;
      float v = fmaxf(x0, x1);
#pragma unroll
      for (int m = 1; m < 16; m <<= 1) v = fmaxf(v, __shfl_xor(v, m, 32));
      const float mn    = fmaxf(m8[g], v);
      const float alpha = __expf(m8[g] - mn);
      m8[g] = mn;
      x0 = __expf(x0 - mn);
      x1 = __expf(x1 - mn);
      float rs = x0 + x1;
#pragma unroll
      for (int m = 1; m < 16; m <<= 1) rs += __shfl_xor(rs, m, 32);
      l8[g] = l8[g] * alpha + rs;
#pragma unroll
      for (int t = 0; t < 4; ++t) o[t][g] *= alpha;
      s0[g] = x0; s1[g] = x1;
    }

    __bf16* pT = &ldsP[wave][0];
#pragma unroll
    for (int g = 0; g < 8; ++g) {
      const int r = g + rofs;
      pT[r * LDP + colk]      = (__bf16)s0[g];
      pT[r * LDP + 16 + colk] = (__bf16)s1[g];
    }
    asm volatile("s_wait_dscnt 0" ::: "memory");
    v16bf pF = frag_ld(&pT[lrow * LDP + lkof]);

#pragma unroll
    for (int t = 0; t < 4; ++t) {
      v16bf bV = frag_ld(&ldsVt[(t * 16 + lrow) * LDV + lkof]);
      o[t] = wmma_bf16(pF, bV, o[t]);
    }
    __syncthreads();
  }

#pragma unroll
  for (int g = 0; g < 8; ++g) {
    const float inv = 1.0f / l8[g];
    const int q = qt + g + rofs;
#pragma unroll
    for (int t = 0; t < 4; ++t) {
      const int e = t * 16 + (lane & 15);
      attn[((size_t)b * SEQ + q) * DMODEL + h * HDIM + e] =
          (__bf16)(o[t][g] * inv);
    }
  }
}

// ---------------------------------------------------------------------------
// Host launcher
// ---------------------------------------------------------------------------
extern "C" void kernel_launch(void* const* d_in, const int* in_sizes, int n_in,
                              void* d_out, int out_size, void* d_ws, size_t ws_size,
                              hipStream_t stream) {
  (void)in_sizes; (void)n_in; (void)out_size; (void)ws_size;

  const float* x     = (const float*)d_in[0];
  const float* w_qkv = (const float*)d_in[1];
  const float* w_out = (const float*)d_in[2];
  float* out = (float*)d_out;

  char* ws = (char*)d_ws;
  __bf16* xb     = (__bf16*)(ws);                                  // 16 MB
  __bf16* wqkvT  = (__bf16*)(ws + 16777216);                       //  6 MB
  __bf16* woutT  = (__bf16*)(ws + 23068672);                       //  2 MB
  __bf16* qkvb   = (__bf16*)(ws + 25165824);                       // 48 MB
  __bf16* attnb  = (__bf16*)(ws + 75497472);                       // 16 MB

  {
    int n = MTOK * DMODEL;
    k_convert_bf16<<<(n + 255) / 256, 256, 0, stream>>>(x, xb, n);
    int nq = DMODEL * D3;
    k_transpose_bf16<<<(nq + 255) / 256, 256, 0, stream>>>(w_qkv, wqkvT,
                                                           DMODEL, D3);
    int no = DMODEL * DMODEL;
    k_transpose_bf16<<<(no + 255) / 256, 256, 0, stream>>>(w_out, woutT,
                                                           DMODEL, DMODEL);
  }

  {
    dim3 grid(MTOK / 128, D3 / 256);
    k_gemm_bf16<false><<<grid, 256, 0, stream>>>(xb, wqkvT, qkvb,
                                                 MTOK, D3, DMODEL);
  }

  {
    dim3 grid(SEQ / 128, NHEADS, BATCH);
    k_flash_attn<<<grid, 256, 0, stream>>>(qkvb, attnb);
  }

  {
    dim3 grid(MTOK / 128, DMODEL / 256);
    k_gemm_bf16<true><<<grid, 256, 0, stream>>>(attnb, woutT, out,
                                                MTOK, DMODEL, DMODEL);
  }
}
